// SACAC_2156073582814
// MI455X (gfx1250) — matrix operand order repaired and checked by reading, not compiled, not run on hardware
//
#include <hip/hip_runtime.h>
#include <math.h>

typedef __attribute__((ext_vector_type(16))) _Float16 v16h;
typedef __attribute__((ext_vector_type(8)))  _Float16 v8h;
typedef __attribute__((ext_vector_type(8)))  float    v8f;

#define BATCH 4
#define CCH   256
#define NPIX  4096            // 64*64
#define BCN   ((size_t)BATCH * CCH * NPIX)
#define L2E   1.44269504088896340736f

// xor-shuffle within the wave via ds_swizzle (group-of-32 mode: [14:10]=xor, [4:0]=and)
template <int M>
__device__ __forceinline__ float swz_xor(float x) {
    return __int_as_float(__builtin_amdgcn_ds_swizzle(__float_as_int(x), (M << 10) | 0x1f));
}

// ---------------------------------------------------------------------------
// Kernel P: pack all projection weights into f16 Wh[320][256] (+ f32 biases).
// Q rows (o<32) are pre-scaled by log2(e) so softmax can use native exp2.
// ---------------------------------------------------------------------------
__global__ __launch_bounds__(256) void prep_kernel(
    const float* __restrict__ qw, const float* __restrict__ qb,
    const float* __restrict__ kw, const float* __restrict__ kb,
    const float* __restrict__ vw, const float* __restrict__ vb,
    _Float16* __restrict__ Wh, float* __restrict__ Bs)
{
    const int idx = blockIdx.x * 256 + threadIdx.x;   // < 320*256
    const int o = idx >> 8, c = idx & 255;
    float w;
    if (o < 32)      w = qw[o * 256 + c] * L2E;
    else if (o < 64) w = kw[(o - 32) * 256 + c];
    else             w = vw[(o - 64) * 256 + c];
    Wh[idx] = (_Float16)w;
    if (idx < 320) {
        float bsv;
        if (idx < 32)      bsv = qb[idx] * L2E;
        else if (idx < 64) bsv = kb[idx - 32];
        else               bsv = vb[idx - 64];
        Bs[idx] = bsv;
    }
}

// ---------------------------------------------------------------------------
// Kernel A: per-pixel channel mean & max for x1 and x2 -> feats [B][2][N]
// ---------------------------------------------------------------------------
__global__ __launch_bounds__(256) void feats_kernel(
    const float* __restrict__ x1, const float* __restrict__ x2,
    float* __restrict__ f1, float* __restrict__ f2)
{
    const int idx = blockIdx.x * 256 + threadIdx.x;   // 0..B*N-1
    const int b = idx >> 12, n = idx & 4095;
    const size_t base = ((size_t)b * CCH) * NPIX + n;
    float s1 = 0.f, m1 = -1e30f, s2 = 0.f, m2 = -1e30f;
    for (int c = 0; c < CCH; ++c) {
        float a = x1[base + (size_t)c * NPIX];
        float bb = x2[base + (size_t)c * NPIX];
        s1 += a;  m1 = fmaxf(m1, a);
        s2 += bb; m2 = fmaxf(m2, bb);
    }
    const size_t fb = ((size_t)b * 2) * NPIX + n;
    f1[fb] = s1 * (1.f / 256.f); f1[fb + NPIX] = m1;
    f2[fb] = s2 * (1.f / 256.f); f2[fb + NPIX] = m2;
}

// ---------------------------------------------------------------------------
// Kernel B: 3x3 conv (SAME) over feats, sigmoid, scale.
// Writes f32 x1s/x2s into d_out (residual + second output) and f16
// pixel-major copies Xh1/Xh2 [b][n][256] for the WMMA projection GEMM.
// ---------------------------------------------------------------------------
__global__ __launch_bounds__(256) void sa_kernel(
    const float* __restrict__ x1, const float* __restrict__ x2,
    const float* __restrict__ f1, const float* __restrict__ f2,
    const float* __restrict__ saw, float* __restrict__ out,
    _Float16* __restrict__ Xh1, _Float16* __restrict__ Xh2)
{
    const int idx = blockIdx.x * 256 + threadIdx.x;
    const int b = idx >> 12, n = idx & 4095;
    const int h = n >> 6, w = n & 63;
    float a1 = 0.f, a2 = 0.f;
    #pragma unroll
    for (int i = 0; i < 2; ++i) {
        #pragma unroll
        for (int kh = 0; kh < 3; ++kh) {
            #pragma unroll
            for (int kw = 0; kw < 3; ++kw) {
                const int hh = h + kh - 1, ww = w + kw - 1;
                if (hh >= 0 && hh < 64 && ww >= 0 && ww < 64) {
                    const float wt = saw[i * 9 + kh * 3 + kw];
                    const size_t fo = ((size_t)b * 2 + i) * NPIX + hh * 64 + ww;
                    a1 += f1[fo] * wt;
                    a2 += f2[fo] * wt;
                }
            }
        }
    }
    const float g1 = 1.f / (1.f + __expf(-a1));
    const float g2 = 1.f / (1.f + __expf(-a2));
    const size_t base = ((size_t)b * CCH) * NPIX + n;
    const size_t xb = ((size_t)(b * NPIX + n)) * CCH;
    for (int c = 0; c < CCH; ++c) {
        const size_t off = base + (size_t)c * NPIX;
        const float v1 = x1[off] * g1;
        const float v2 = x2[off] * g2;
        out[off]       = v1;               // x1s (also residual for out1)
        out[BCN + off] = v2;               // x2s (second output, final)
        Xh1[xb + c] = (_Float16)v1;
        Xh2[xb + c] = (_Float16)v2;
    }
}

// ---------------------------------------------------------------------------
// Kernel C: WMMA projection GEMM.  D'[pixel][o] = X^T * W^T (+bias).
// One wave owns a 16-pixel tile and sweeps all 20 output-channel tiles
// (og 0-1: Q from x1s; og 2-3: K from x2s; og 4-19: V from x2s),
// accumulating over 8 k-tiles of 32 input channels -> 160 WMMAs/wave.
//   A-tile: Xh [b][n][256] rows (pixel-major)  -> two b128 loads per lane
//   B-tile: Wh [o][256] row-major -> one v16h load per lane (contiguous k = 16*kh + s, as A)
//   V epilogue: rows are pixels -> one packed v8h store per lane per og.
// ---------------------------------------------------------------------------
__global__ __launch_bounds__(128) void proj_kernel(
    const _Float16* __restrict__ Xh1, const _Float16* __restrict__ Xh2,
    const _Float16* __restrict__ Wh,  const float* __restrict__ Bs,
    _Float16* __restrict__ Qh, _Float16* __restrict__ Kh, _Float16* __restrict__ Vh)
{
    const int lane = threadIdx.x & 31;
    const int wv   = threadIdx.x >> 5;
    const int gw   = blockIdx.x * 4 + wv;       // 0..1023
    const int b    = gw >> 8;
    const int n0   = (gw & 255) << 4;
    const int j16  = lane & 15;
    const int kh   = lane >> 4;

    // resident A-tiles for both sources (lane = pixel row j16)
    v16h A1[8], A2[8];
    #pragma unroll
    for (int ct = 0; ct < 8; ++ct) {
        const _Float16* r1 = Xh1 + ((size_t)(b * NPIX + n0 + j16)) * CCH + ct * 32;
        const _Float16* r2 = Xh2 + ((size_t)(b * NPIX + n0 + j16)) * CCH + ct * 32;
        v8h lo1 = *(const v8h*)(r1 + kh * 16), hi1 = *(const v8h*)(r1 + kh * 16 + 8);
        v8h lo2 = *(const v8h*)(r2 + kh * 16), hi2 = *(const v8h*)(r2 + kh * 16 + 8);
        #pragma unroll
        for (int e = 0; e < 8; ++e) {
            A1[ct][e] = lo1[e]; A1[ct][8 + e] = hi1[e];
            A2[ct][e] = lo2[e]; A2[ct][8 + e] = hi2[e];
        }
    }

    for (int og = 0; og < 20; ++og) {
        const int o = og * 16 + j16;            // this lane's output channel (col)
        const _Float16* Wt = Wh + (size_t)o * CCH + kh * 16;
        v8f acc = {};
        if (og < 2) {
            #pragma unroll
            for (int ct = 0; ct < 8; ++ct) {
                v16h bt = *(const v16h*)(Wt + ct * 32);
                acc = __builtin_amdgcn_wmma_f32_16x16x32_f16(false, A1[ct], false, bt,
                                                             (short)0, acc, false, false);
            }
        } else {
            #pragma unroll
            for (int ct = 0; ct < 8; ++ct) {
                v16h bt = *(const v16h*)(Wt + ct * 32);
                acc = __builtin_amdgcn_wmma_f32_16x16x32_f16(false, A2[ct], false, bt,
                                                             (short)0, acc, false, false);
            }
        }
        const float bias = Bs[o];
        #pragma unroll
        for (int v = 0; v < 8; ++v) acc[v] += bias;

        if (og < 4) {                            // Q or K -> [b][n][32]
            _Float16* dst = (og < 2) ? Qh : Kh;
            const int oo = (og < 2) ? og * 16 + j16 : (og - 2) * 16 + j16;
            #pragma unroll
            for (int v = 0; v < 8; ++v)
                dst[((size_t)(b * NPIX + n0 + (kh << 3) + v)) * 32 + oo] = (_Float16)acc[v];
        } else {                                 // V -> [b][c][n], 8 consecutive pixels
            const int c = (og - 4) * 16 + j16;
            v8h hh;
            #pragma unroll
            for (int v = 0; v < 8; ++v) hh[v] = (_Float16)acc[v];
            *(v8h*)(Vh + ((size_t)(b * CCH + c)) * NPIX + n0 + (kh << 3)) = hh;
        }
    }
}

// ---------------------------------------------------------------------------
// Kernel D: flash attention.  One wave = 16 query rows; 4 waves/block.
// S' = (log2e-scaled Q) K^T via v_wmma_f32_16x16x32_f16, online softmax in
// exp2 domain, O += P V (16 WMMAs per 32-key tile), out1 = gamma*O/sum + x1s.
// ---------------------------------------------------------------------------
__global__ __launch_bounds__(128) void attn_kernel(
    const _Float16* __restrict__ Qh, const _Float16* __restrict__ Kh,
    const _Float16* __restrict__ Vh, const float* __restrict__ gm,
    float* __restrict__ out)
{
    __shared__ _Float16 sP[4][16 * 32];             // per-wave P tile (1KB each)
    const int lane = threadIdx.x & 31;
    const int wv   = threadIdx.x >> 5;
    const int b    = blockIdx.x >> 6;               // 64 blocks per batch
    const int qt   = ((blockIdx.x & 63) << 2) | wv; // 0..255
    const int n0   = qt << 4;
    const int j16  = lane & 15;
    const int kh   = lane >> 4;
    const float gamma = gm[0];

    // ---- A-tile: Q rows n0..n0+15, lane holds row j16, K-dims contiguous (element s = dim 16*kh + s)
    v16h qa;
    {
        const _Float16* Qrow = Qh + ((size_t)(b * NPIX + n0 + j16)) * 32;
        v8h lo = *(const v8h*)(Qrow + kh * 16);         // K = kh*16 + 0..7
        v8h hi = *(const v8h*)(Qrow + kh * 16 + 8);     // K = kh*16 + 8..15
        #pragma unroll
        for (int e = 0; e < 8; ++e) { qa[e] = lo[e]; qa[8 + e] = hi[e]; }
    }

    const _Float16* Kb = Kh + ((size_t)b * NPIX) * 32;
    const _Float16* Vb = Vh + ((size_t)b * CCH) * NPIX;

    v8f zero8 = {};
    v8f Oacc[16];
    #pragma unroll
    for (int cg = 0; cg < 16; ++cg) Oacc[cg] = zero8;
    float rmax[8], rsum[8];
    #pragma unroll
    for (int v = 0; v < 8; ++v) { rmax[v] = -1e30f; rsum[v] = 0.f; }

    for (int j0 = 0; j0 < NPIX; j0 += 32) {
        // B-tiles of K^T: lane = key column j16, 16 contiguous dims
        v16h kb0 = *(const v16h*)(Kb + ((size_t)(j0 + j16)) * 32 + kh * 16);
        v16h kb1 = *(const v16h*)(Kb + ((size_t)(j0 + 16 + j16)) * 32 + kh * 16);
        v8f s0 = __builtin_amdgcn_wmma_f32_16x16x32_f16(false, qa, false, kb0,
                                                        (short)0, zero8, false, false);
        v8f s1 = __builtin_amdgcn_wmma_f32_16x16x32_f16(false, qa, false, kb1,
                                                        (short)0, zero8, false, false);
        // ---- online softmax per query row r = v + 8*kh (exp2 domain)
        float corr[8];
        __syncthreads();                      // WAR guard on sP vs previous loads
        #pragma unroll
        for (int v = 0; v < 8; ++v) {
            float t = fmaxf(s0[v], s1[v]);
            t = fmaxf(t, swz_xor<1>(t)); t = fmaxf(t, swz_xor<2>(t));
            t = fmaxf(t, swz_xor<4>(t)); t = fmaxf(t, swz_xor<8>(t));
            const float nm = fmaxf(rmax[v], t);
            const float c_ = exp2f(rmax[v] - nm);
            rmax[v] = nm;
            const float p0 = exp2f(s0[v] - nm);
            const float p1 = exp2f(s1[v] - nm);
            float rs = p0 + p1;
            rs += swz_xor<1>(rs); rs += swz_xor<2>(rs);
            rs += swz_xor<4>(rs); rs += swz_xor<8>(rs);
            rsum[v] = rsum[v] * c_ + rs;
            corr[v] = c_;
            const int r = (kh << 3) + v;      // D-layout row
            sP[wv][r * 32 + j16]      = (_Float16)p0;
            sP[wv][r * 32 + 16 + j16] = (_Float16)p1;
        }
        #pragma unroll
        for (int cg = 0; cg < 16; ++cg) {
            #pragma unroll
            for (int v = 0; v < 8; ++v) Oacc[cg][v] *= corr[v];
        }
        __syncthreads();
        // ---- re-gather P in A-layout (16x32 f16)
        v16h pa;
        {
            const _Float16* Pr = &sP[wv][j16 * 32];
            v8h lo = *(const v8h*)(Pr + kh * 16);
            v8h hi = *(const v8h*)(Pr + kh * 16 + 8);
            #pragma unroll
            for (int e = 0; e < 8; ++e) { pa[e] = lo[e]; pa[8 + e] = hi[e]; }
        }
        // ---- O += P * V over 16 channel groups
        #pragma unroll
        for (int cg = 0; cg < 16; ++cg) {
            v16h vbv = *(const v16h*)(Vb + ((size_t)(cg * 16 + j16)) * NPIX + j0 + kh * 16);
            Oacc[cg] = __builtin_amdgcn_wmma_f32_16x16x32_f16(false, pa, false, vbv,
                                                              (short)0, Oacc[cg], false, false);
        }
    }

    // ---- out1 = gamma * O / rowsum + x1s  (x1s currently lives in out[])
    float gscale[8];
    #pragma unroll
    for (int v = 0; v < 8; ++v) gscale[v] = gamma / rsum[v];
    const size_t obase = ((size_t)b * CCH) * NPIX;
    #pragma unroll
    for (int cg = 0; cg < 16; ++cg) {
        const int ch = cg * 16 + j16;
        #pragma unroll
        for (int v = 0; v < 8; ++v) {
            const int r = (kh << 3) + v;
            const size_t idx = obase + (size_t)ch * NPIX + n0 + r;
            out[idx] = Oacc[cg][v] * gscale[v] + out[idx];
        }
    }
}

// ---------------------------------------------------------------------------
extern "C" void kernel_launch(void* const* d_in, const int* in_sizes, int n_in,
                              void* d_out, int out_size, void* d_ws, size_t ws_size,
                              hipStream_t stream)
{
    (void)in_sizes; (void)n_in; (void)out_size; (void)ws_size;
    const float* x1  = (const float*)d_in[0];
    const float* x2  = (const float*)d_in[1];
    const float* saw = (const float*)d_in[2];
    const float* qw  = (const float*)d_in[3];
    const float* qb  = (const float*)d_in[4];
    const float* kw  = (const float*)d_in[5];
    const float* kb  = (const float*)d_in[6];
    const float* vw  = (const float*)d_in[7];
    const float* vb  = (const float*)d_in[8];
    const float* gm  = (const float*)d_in[9];
    float* out = (float*)d_out;

    char* ws = (char*)d_ws;
    const size_t MB = 1u << 20;
    float*    f1  = (float*)ws;                                // 128 KB
    float*    f2  = (float*)(ws + (128 << 10));                // 128 KB
    _Float16* Qh  = (_Float16*)(ws + (256 << 10));             // 1 MB
    _Float16* Kh  = (_Float16*)(ws + (256 << 10) + 1 * MB);    // 1 MB
    _Float16* Vh  = (_Float16*)(ws + (256 << 10) + 2 * MB);    // 8 MB
    _Float16* Xh1 = (_Float16*)(ws + (256 << 10) + 10 * MB);   // 8 MB
    _Float16* Xh2 = (_Float16*)(ws + (256 << 10) + 18 * MB);   // 8 MB
    _Float16* Wh  = (_Float16*)(ws + (256 << 10) + 26 * MB);   // 160 KB
    float*    Bs  = (float*)(ws + (256 << 10) + 26 * MB + (160 << 10)); // 1.25 KB

    prep_kernel <<<320, 256, 0, stream>>>(qw, qb, kw, kb, vw, vb, Wh, Bs);
    feats_kernel<<<64, 256, 0, stream>>>(x1, x2, f1, f2);
    sa_kernel   <<<64, 256, 0, stream>>>(x1, x2, f1, f2, saw, out, Xh1, Xh2);
    proj_kernel <<<256, 128, 0, stream>>>(Xh1, Xh2, Wh, Bs, Qh, Kh, Vh);
    attn_kernel <<<BATCH * 64, 128, 0, stream>>>(Qh, Kh, Vh, gm, out);
}
